// LTransformerDecoder_53687091200456
// MI455X (gfx1250) — matrix operand order repaired and checked by reading, not compiled, not run on hardware
//
#include <hip/hip_runtime.h>
#include <hip/hip_bf16.h>
#include <math.h>

// ---------------------------------------------------------------------------
// LTransformer decoder forward for gfx1250 (MI455X, wave32, WMMA).
// B=4 S=1024 V=32000 D=1040 H=16 HD=65 HS=64 L=4.  M = B*S = 4096 rows.
// GEMMs run through v_wmma_f32_16x16x32_bf16 (fp32 weights/activations are
// converted to bf16 in LDS; fp32 accumulate). Everything else is row kernels.
// ---------------------------------------------------------------------------

#define LT_B   4
#define LT_S   1024
#define LT_V   32000
#define LT_D   1040
#define LT_H   16
#define LT_HD  65
#define LT_HS  64
#define LT_L   4
#define LT_M   (LT_B * LT_S)          // 4096
#define LT_EPS 1e-6f

typedef __attribute__((ext_vector_type(16))) __bf16 v16bf;
typedef __attribute__((ext_vector_type(2)))  __bf16 bf16x2;
typedef __attribute__((ext_vector_type(8)))  float  v8f;

// ---------------------------------------------------------------------------
// Block-wide sum over 256 threads (8 waves).
// ---------------------------------------------------------------------------
__device__ __forceinline__ float lt_block_sum(float v) {
  __shared__ float red[8];
  int lane = threadIdx.x & 31, wid = threadIdx.x >> 5;
#pragma unroll
  for (int o = 16; o; o >>= 1) v += __shfl_xor(v, o, 32);
  __syncthreads();               // protect red[] from a previous call's readers
  if (lane == 0) red[wid] = v;
  __syncthreads();
  float t = 0.f;
#pragma unroll
  for (int i = 0; i < 8; ++i) t += red[i];
  return t;
}

__device__ __forceinline__ float lt_gelu(float x) {
  const float c0 = 0.7978845608028654f;   // sqrt(2/pi)
  return 0.5f * x * (1.f + tanhf(c0 * (x + 0.044715f * x * x * x)));
}

// ---------------------------------------------------------------------------
// Tiled bf16-WMMA GEMM:  C[M,N] = act(A[M,K] @ W[K,N] + bias)
// Block tile 128x128, K-tile 32. 256 threads = 8 waves (2x4); each wave owns
// a 64x32 patch = 4x2 WMMA 16x16 accumulators.
//
// Software-pipelined staging: tile k+1's global loads (registers) are issued
// before the WMMAs of tile k, so the s_wait_loadcnt lands in the next LDS
// store phase and global latency hides under the matrix ops. Per-thread
// addresses are separable (base + j*stride) so guards hoist out of the loops.
// M must be a multiple of 128 (always 4096 here). K and N are arbitrary.
// ---------------------------------------------------------------------------
__global__ __launch_bounds__(256) void lt_gemm_wmma(
    const float* __restrict__ A, const float* __restrict__ W,
    const float* __restrict__ bias, float* __restrict__ C,
    int Ndim, int Kdim, int act) {
  __shared__ __bf16 As[128][40];   // [m][k], padded (80 B rows)
  __shared__ __bf16 Bs[128][40];   // [n][k] (transposed), padded

  const int tid  = threadIdx.x;
  const int lane = tid & 31, wid = tid >> 5;
  const int m0 = blockIdx.y * 128;
  const int n0 = blockIdx.x * 128;
  const int wm = (wid >> 2) * 64;       // wave M offset within block: 0/64
  const int wn = (wid & 3) * 32;        // wave N offset within block: 0..96

  v8f acc[4][2];
#pragma unroll
  for (int mt = 0; mt < 4; ++mt)
#pragma unroll
    for (int nt = 0; nt < 2; ++nt)
#pragma unroll
      for (int r = 0; r < 8; ++r) acc[mt][nt][r] = 0.f;

  // ---- per-thread staging descriptors (constant across k-tiles) ----
  // A: pair j covers row (tid>>4)+16j, k-pair (tid&15)*2  -> stride 16 rows.
  // B: pair j covers n = tid&127,  k-pair (tid>>7)*2 + 4j -> stride 4 rows.
  const int arow0 = tid >> 4;            // 0..15
  const int akp   = (tid & 15) << 1;     // 0,2,..,30
  const int bn    = tid & 127;           // 0..127
  const int bkp0  = (tid >> 7) << 1;     // 0 or 2
  const int gn    = n0 + bn;
  const bool nv   = gn < Ndim;
  const int gnc   = nv ? gn : 0;
  const float* arow = A + (size_t)(m0 + arow0) * Kdim;

  float fa[8], fb[8], ga[8], gb[8];

  auto load_tile = [&](int k0) {
    int ka = k0 + akp;
    int ia = (ka < Kdim) ? ka : 0;
    int ib = (ka + 1 < Kdim) ? ka + 1 : 0;
#pragma unroll
    for (int j = 0; j < 8; ++j) {
      size_t roff = (size_t)16 * j * Kdim;
      fa[j] = arow[roff + ia];
      fb[j] = arow[roff + ib];
    }
#pragma unroll
    for (int j = 0; j < 8; ++j) {
      int kb = k0 + bkp0 + 4 * j;
      int ja = (kb < Kdim) ? kb : 0;
      int jb = (kb + 1 < Kdim) ? kb + 1 : 0;
      ga[j] = W[(size_t)ja * Ndim + gnc];
      gb[j] = W[(size_t)jb * Ndim + gnc];
    }
  };

  auto store_tile = [&](int k0) {
    int ka = k0 + akp;
    bool va = ka < Kdim, vb = ka + 1 < Kdim;
#pragma unroll
    for (int j = 0; j < 8; ++j) {
      bf16x2 pk;
      pk.x = (__bf16)(va ? fa[j] : 0.f);
      pk.y = (__bf16)(vb ? fb[j] : 0.f);
      *(bf16x2*)&As[arow0 + 16 * j][akp] = pk;   // immediate LDS offsets
    }
#pragma unroll
    for (int j = 0; j < 8; ++j) {
      int kb = k0 + bkp0 + 4 * j;
      bool wa = nv && (kb < Kdim), wb = nv && (kb + 1 < Kdim);
      bf16x2 pk;
      pk.x = (__bf16)(wa ? ga[j] : 0.f);
      pk.y = (__bf16)(wb ? gb[j] : 0.f);
      *(bf16x2*)&Bs[bn][bkp0 + 4 * j] = pk;      // immediate LDS offsets
    }
  };

  const int lm    = lane & 15;
  const int abase = (lane < 16) ? 0 : 8;    // A K-interleave (ISA 7.12.2)
  const int khalf = (lane < 16) ? 0 : 16;   // unused; B uses abase order like A

  load_tile(0);                              // pipeline prologue

  for (int k0 = 0; k0 < Kdim; k0 += 32) {
    __syncthreads();                         // LDS free from previous tile
    store_tile(k0);                          // waits on this tile's loads
    __syncthreads();

    if (k0 + 32 < Kdim) load_tile(k0 + 32);  // issue next tile's loads now
    if (k0 + 64 < Kdim) {                    // prefetch tile after next
      __builtin_prefetch(arow + k0 + 64, 0, 1);
      int pr = k0 + 64 + (tid >> 3);
      if (pr < Kdim)
        __builtin_prefetch(&W[(size_t)pr * Ndim + n0 + ((tid & 7) << 4)], 0, 1);
    }

    // ---- load fragments per ISA VGPR layouts (merge to ds_load_b128) ----
    v16bf afrag[4], bfrag[2];
#pragma unroll
    for (int mt = 0; mt < 4; ++mt) {
      const __bf16* ar = &As[wm + mt * 16 + lm][0];
#pragma unroll
      for (int j = 0; j < 8; ++j) {
        afrag[mt][j]     = ar[abase + j];
        afrag[mt][8 + j] = ar[16 + abase + j];
      }
    }
#pragma unroll
    for (int nt = 0; nt < 2; ++nt) {
      const __bf16* br = &Bs[wn + nt * 16 + lm][0];
#pragma unroll
      for (int j = 0; j < 8; ++j) {
        bfrag[nt][j]     = br[abase + j];
        bfrag[nt][8 + j] = br[16 + abase + j];
      }
    }

    // ---- 8 WMMAs per wave per K-tile (globals for k+1 still in flight) ----
#pragma unroll
    for (int mt = 0; mt < 4; ++mt)
#pragma unroll
      for (int nt = 0; nt < 2; ++nt)
        acc[mt][nt] = __builtin_amdgcn_wmma_f32_16x16x32_bf16(
            false, afrag[mt], false, bfrag[nt], (short)0, acc[mt][nt],
            false, false);
  }

  // ---- epilogue: bias + optional GELU, guarded N stores ----
  const int mo = (lane >> 4) << 3;   // C/D: m = r + 8*(lane>=16)
#pragma unroll
  for (int nt = 0; nt < 2; ++nt) {
    int n = n0 + wn + nt * 16 + lm;
    if (n >= Ndim) continue;
    float bv = bias ? bias[n] : 0.f;
#pragma unroll
    for (int mt = 0; mt < 4; ++mt) {
#pragma unroll
      for (int r = 0; r < 8; ++r) {
        int m = m0 + wm + mt * 16 + r + mo;
        float v = acc[mt][nt][r] + bv;
        if (act == 1) v = lt_gelu(v);
        C[(size_t)m * Ndim + n] = v;
      }
    }
  }
}

// ---------------------------------------------------------------------------
// Embedding gather + add_time: x[row][0] = sqrt(1+sum xs^2), x[row][1+i]=xs[i]
// ---------------------------------------------------------------------------
__global__ __launch_bounds__(256) void lt_embed(
    const int* __restrict__ tok, const float* __restrict__ tab,
    float* __restrict__ x) {
  int row = blockIdx.x;
  const float* e = tab + (size_t)tok[row] * (LT_D - 1);
  float s = 0.f;
  for (int i = threadIdx.x; i < LT_D - 1; i += 256) { float v = e[i]; s += v * v; }
  s = lt_block_sum(s);
  float* xr = x + (size_t)row * LT_D;
  if (threadIdx.x == 0) xr[0] = sqrtf(1.f + s);
  for (int i = threadIdx.x; i < LT_D - 1; i += 256) xr[1 + i] = e[i];
}

// ---------------------------------------------------------------------------
// l_rmsnorm: out[0]=sqrt(1+sum y^2), out[1+i]=xs_i*rsqrt(mean(xs^2)+eps)*w_i
// ---------------------------------------------------------------------------
__global__ __launch_bounds__(256) void lt_rmsnorm(
    const float* __restrict__ x, const float* __restrict__ w,
    float* __restrict__ out) {
  int row = blockIdx.x;
  const float* xr = x + (size_t)row * LT_D;
  float s1 = 0.f, s2 = 0.f;
  for (int i = threadIdx.x; i < LT_D - 1; i += 256) {
    float xs = xr[1 + i], yw = xs * w[i];
    s1 += xs * xs; s2 += yw * yw;
  }
  s1 = lt_block_sum(s1);
  s2 = lt_block_sum(s2);
  float scale = rsqrtf(s1 / (float)(LT_D - 1) + LT_EPS);
  float* orow = out + (size_t)row * LT_D;
  if (threadIdx.x == 0) orow[0] = sqrtf(1.f + scale * scale * s2);
  for (int i = threadIdx.x; i < LT_D - 1; i += 256)
    orow[1 + i] = xr[1 + i] * scale * w[i];
}

// ---------------------------------------------------------------------------
// RoPE + add_time + head transpose: in (b*S+s, h*64+d) -> out ((b*H+h)*S+s, 65)
// One wave per (row, head); lane = rotation pair.
// ---------------------------------------------------------------------------
__global__ __launch_bounds__(256) void lt_rope_addtime(
    const float* __restrict__ in, float* __restrict__ out) {
  int w = blockIdx.x * 8 + (threadIdx.x >> 5);
  int lane = threadIdx.x & 31;
  int h = w & (LT_H - 1), row = w >> 4;
  int b = row >> 10, s = row & (LT_S - 1);
  const float* src = in + (size_t)row * (LT_H * LT_HS) + h * LT_HS;
  float x1 = src[2 * lane], x2 = src[2 * lane + 1];
  float inv = __expf(-((float)(2 * lane) / (float)LT_HS) * 9.210340371976184f);
  float ang = (float)s * inv;
  float sn, cs; __sincosf(ang, &sn, &cs);
  float r1 = x1 * cs - x2 * sn;
  float r2 = x1 * sn + x2 * cs;
  float sq = r1 * r1 + r2 * r2;
#pragma unroll
  for (int o = 16; o; o >>= 1) sq += __shfl_xor(sq, o, 32);
  float* dst = out + ((size_t)(b * LT_H + h) * LT_S + s) * LT_HD;
  if (lane == 0) dst[0] = sqrtf(1.f + sq);
  dst[1 + 2 * lane] = r1;
  dst[2 + 2 * lane] = r2;
}

// add_time + head transpose for V (no rope)
__global__ __launch_bounds__(256) void lt_v_addtime(
    const float* __restrict__ in, float* __restrict__ out) {
  int w = blockIdx.x * 8 + (threadIdx.x >> 5);
  int lane = threadIdx.x & 31;
  int h = w & (LT_H - 1), row = w >> 4;
  int b = row >> 10, s = row & (LT_S - 1);
  const float* src = in + (size_t)row * (LT_H * LT_HS) + h * LT_HS;
  float v1 = src[lane], v2 = src[lane + 32];
  float sq = v1 * v1 + v2 * v2;
#pragma unroll
  for (int o = 16; o; o >>= 1) sq += __shfl_xor(sq, o, 32);
  float* dst = out + ((size_t)(b * LT_H + h) * LT_S + s) * LT_HD;
  if (lane == 0) dst[0] = sqrtf(1.f + sq);
  dst[1 + lane] = v1;
  dst[1 + lane + 32] = v2;
}

// ---------------------------------------------------------------------------
// Causal attention + Minkowski l_normalize, fused.
// grid = (S/8, B*H); one wave per query row; scores staged in LDS.
// score_j = (sum_{d>=1} q_d k_d - q_0 k_0) / sqrt(HD),   j <= qi.
// ---------------------------------------------------------------------------
__global__ __launch_bounds__(256) void lt_attention(
    const float* __restrict__ q, const float* __restrict__ k,
    const float* __restrict__ v, float* __restrict__ m_out) {
  __shared__ float sc[8][LT_S];
  __shared__ float qsh[8][LT_HD];
  int wid = threadIdx.x >> 5, lane = threadIdx.x & 31;
  int bh = blockIdx.y;
  int qi = blockIdx.x * 8 + wid;
  const float* qrow = q + ((size_t)bh * LT_S + qi) * LT_HD;
  for (int d = lane; d < LT_HD; d += 32) qsh[wid][d] = qrow[d];
  __syncthreads();

  const float scale = rsqrtf((float)LT_HD);
  float mx = -3.0e38f;
  for (int j = lane; j <= qi; j += 32) {
    const float* kr = k + ((size_t)bh * LT_S + j) * LT_HD;
    float dv = -qsh[wid][0] * kr[0];
#pragma unroll 8
    for (int d = 1; d < LT_HD; ++d) dv += qsh[wid][d] * kr[d];
    dv *= scale;
    sc[wid][j] = dv;
    mx = fmaxf(mx, dv);
  }
#pragma unroll
  for (int o = 16; o; o >>= 1) mx = fmaxf(mx, __shfl_xor(mx, o, 32));
  float se = 0.f;
  for (int j = lane; j <= qi; j += 32) {
    float e = __expf(sc[wid][j] - mx);
    sc[wid][j] = e; se += e;
  }
#pragma unroll
  for (int o = 16; o; o >>= 1) se += __shfl_xor(se, o, 32);
  float isum = 1.f / se;

  float od[3]; int nd = 0;
  float part = 0.f;
  for (int d = lane; d < LT_HD; d += 32) {
    float a = 0.f;
    for (int j = 0; j <= qi; ++j)
      a += sc[wid][j] * v[((size_t)bh * LT_S + j) * LT_HD + d];
    float o = a * isum;
    od[nd++] = o;
    part += (d == 0) ? o * o : -o * o;
  }
#pragma unroll
  for (int o = 16; o; o >>= 1) part += __shfl_xor(part, o, 32);
  float r = rsqrtf(fmaxf(part, LT_EPS));
  int b = bh >> 4, h = bh & (LT_H - 1);
  float* orow = m_out + ((size_t)(b * LT_S + qi)) * LT_D + h * LT_HD;
  nd = 0;
  for (int d = lane; d < LT_HD; d += 32) orow[d] = od[nd++] * r;
}

// ---------------------------------------------------------------------------
// add_time row kernel: out[row][0]=sqrt(1+sum in^2), out[row][1+i]=in[i]
// ---------------------------------------------------------------------------
__global__ __launch_bounds__(256) void lt_addtime(
    const float* __restrict__ in, float* __restrict__ out, int n) {
  int row = blockIdx.x;
  const float* ir = in + (size_t)row * n;
  float s = 0.f;
  for (int i = threadIdx.x; i < n; i += 256) { float v = ir[i]; s += v * v; }
  s = lt_block_sum(s);
  float* orow = out + (size_t)row * (n + 1);
  if (threadIdx.x == 0) orow[0] = sqrtf(1.f + s);
  for (int i = threadIdx.x; i < n; i += 256) orow[1 + i] = ir[i];
}

// ---------------------------------------------------------------------------
// Fused l_linear time + lresnet:  y = x + [sqrt(1+|a|^2), a]/RS ;
// x_out = y * rsqrt(max(y0^2 - sum_{d>=1} y_d^2, eps)).   In-place safe.
// ---------------------------------------------------------------------------
__global__ __launch_bounds__(256) void lt_linres(
    const float* __restrict__ x, const float* __restrict__ pre,
    float* __restrict__ out, float rs_inv) {
  int row = blockIdx.x;
  const float* xr = x + (size_t)row * LT_D;
  const float* ar = pre + (size_t)row * (LT_D - 1);
  float x0 = xr[0];
  float s2 = 0.f, syq = 0.f;
  for (int i = threadIdx.x; i < LT_D - 1; i += 256) {
    float a = ar[i];
    float y = xr[1 + i] + a * rs_inv;
    s2 += a * a; syq += y * y;
  }
  s2  = lt_block_sum(s2);
  syq = lt_block_sum(syq);
  float y0 = x0 + sqrtf(1.f + s2) * rs_inv;
  float r = rsqrtf(fmaxf(y0 * y0 - syq, LT_EPS));
  float* orow = out + (size_t)row * LT_D;
  if (threadIdx.x == 0) orow[0] = y0 * r;
  for (int i = threadIdx.x; i < LT_D - 1; i += 256)
    orow[1 + i] = (xr[1 + i] + ar[i] * rs_inv) * r;
}

// ---------------------------------------------------------------------------
// Host launch
// ---------------------------------------------------------------------------
static inline void lt_gemm(const float* A, const float* W, const float* bias,
                           float* C, int N, int K, int act, hipStream_t s) {
  dim3 grid((N + 127) / 128, LT_M / 128);
  lt_gemm_wmma<<<grid, 256, 0, s>>>(A, W, bias, C, N, K, act);
}

extern "C" void kernel_launch(void* const* d_in, const int* in_sizes, int n_in,
                              void* d_out, int out_size, void* d_ws, size_t ws_size,
                              hipStream_t stream) {
  const int*   tok  = (const int*)  d_in[0];
  const float* etab = (const float*)d_in[1];
  const float* ln1  = (const float*)d_in[2];
  const float* wq   = (const float*)d_in[3];
  const float* bq   = (const float*)d_in[4];
  const float* wk   = (const float*)d_in[5];
  const float* bk   = (const float*)d_in[6];
  const float* wv   = (const float*)d_in[7];
  const float* bv   = (const float*)d_in[8];
  const float* wo   = (const float*)d_in[9];
  const float* bo   = (const float*)d_in[10];
  const float* ln2  = (const float*)d_in[11];
  const float* w1   = (const float*)d_in[12];
  const float* b1   = (const float*)d_in[13];
  const float* w2   = (const float*)d_in[14];
  const float* b2   = (const float*)d_in[15];
  const float* lnf  = (const float*)d_in[16];
  const float* wprj = (const float*)d_in[17];
  const float* bprj = (const float*)d_in[18];
  const float* wmap = (const float*)d_in[19];
  const float* bmap = (const float*)d_in[20];

  const float rs_inv = 1.f / (2.f * sqrtf((float)LT_D));

  float* ws = (float*)d_ws;
  size_t o = 0;
  auto take = [&](size_t n) { float* p = ws + o; o += n; return p; };
  float* X   = take((size_t)LT_M * LT_D);
  float* T1  = take((size_t)LT_M * LT_D);
  float* QH  = take((size_t)LT_M * LT_H * LT_HS);
  float* KH  = take((size_t)LT_M * LT_H * LT_HS);
  float* VH  = take((size_t)LT_M * LT_H * LT_HS);
  float* Qb  = take((size_t)LT_B * LT_H * LT_S * LT_HD);
  float* Kb  = take((size_t)LT_B * LT_H * LT_S * LT_HD);
  float* Vb  = take((size_t)LT_B * LT_H * LT_S * LT_HD);
  float* MM  = take((size_t)LT_M * LT_D);
  float* PRE = take((size_t)LT_M * (LT_D - 1));
  float* G   = take((size_t)LT_M * 4 * LT_D);
  float* H2  = take((size_t)LT_M * (4 * LT_D + 1));
  (void)ws_size; (void)n_in; (void)in_sizes; (void)out_size;

  lt_embed<<<LT_M, 256, 0, stream>>>(tok, etab, X);

  for (int l = 0; l < LT_L; ++l) {
    const float* wql = wq + (size_t)l * LT_D * (LT_H * LT_HS);
    const float* wkl = wk + (size_t)l * LT_D * (LT_H * LT_HS);
    const float* wvl = wv + (size_t)l * LT_D * (LT_H * LT_HS);
    const float* wol = wo + (size_t)l * LT_D * (LT_D - 1);
    const float* w1l = w1 + (size_t)l * LT_D * (4 * LT_D);
    const float* w2l = w2 + (size_t)l * (4 * LT_D + 1) * (LT_D - 1);

    // lx = l_rmsnorm(x, ln1)
    lt_rmsnorm<<<LT_M, 256, 0, stream>>>(X, ln1 + (size_t)l * (LT_D - 1), T1);

    // QKV projections (WMMA)
    lt_gemm(T1, wql, bq + (size_t)l * LT_H * LT_HS, QH, LT_H * LT_HS, LT_D, 0, stream);
    lt_gemm(T1, wkl, bk + (size_t)l * LT_H * LT_HS, KH, LT_H * LT_HS, LT_D, 0, stream);
    lt_gemm(T1, wvl, bv + (size_t)l * LT_H * LT_HS, VH, LT_H * LT_HS, LT_D, 0, stream);

    // rope + add_time + transpose to (B,H,S,HD)
    lt_rope_addtime<<<LT_M * LT_H / 8, 256, 0, stream>>>(QH, Qb);
    lt_rope_addtime<<<LT_M * LT_H / 8, 256, 0, stream>>>(KH, Kb);
    lt_v_addtime  <<<LT_M * LT_H / 8, 256, 0, stream>>>(VH, Vb);

    // attention + l_normalize -> m (B,S,H*HD)
    lt_attention<<<dim3(LT_S / 8, LT_B * LT_H), 256, 0, stream>>>(Qb, Kb, Vb, MM);

    // ax = l_linear(m, wo, bo); x = lresnet(x, ax)
    lt_gemm(MM, wol, bo + (size_t)l * (LT_D - 1), PRE, LT_D - 1, LT_D, 0, stream);
    lt_linres<<<LT_M, 256, 0, stream>>>(X, PRE, X, rs_inv);

    // FFN
    lt_rmsnorm<<<LT_M, 256, 0, stream>>>(X, ln2 + (size_t)l * (LT_D - 1), T1);
    lt_gemm(T1, w1l, b1 + (size_t)l * 4 * LT_D, G, 4 * LT_D, LT_D, /*gelu*/1, stream);
    lt_addtime<<<LT_M, 256, 0, stream>>>(G, H2, 4 * LT_D);
    lt_gemm(H2, w2l, b2 + (size_t)l * (LT_D - 1), PRE, LT_D - 1, 4 * LT_D + 1, 0, stream);
    lt_linres<<<LT_M, 256, 0, stream>>>(X, PRE, X, rs_inv);
  }

  // x = l_linear(x, wproj, bproj)
  lt_gemm(X, wprj, bprj, PRE, LT_D - 1, LT_D, 0, stream);
  float* P = MM;                                  // reuse (dead after layers)
  lt_addtime<<<LT_M, 256, 0, stream>>>(PRE, P, LT_D - 1);
  // x = l_rmsnorm(x, lnf)
  lt_rmsnorm<<<LT_M, 256, 0, stream>>>(P, lnf, T1);
  // out = l_linear(x, wmap, bmap)[..., 1:]  ==  x @ wmap + bmap
  lt_gemm(T1, wmap, bmap, (float*)d_out, LT_V, LT_D, 0, stream);
}
